// UGCALayer_34935263986215
// MI455X (gfx1250) — matrix operand order repaired and checked by reading, not compiled, not run on hardware
//
#include <hip/hip_runtime.h>
#include <math.h>

// ---------------------------------------------------------------------------
// UGCA layer for MI455X (gfx1250, wave32, WMMA).
// All GEMMs run through v_wmma_f32_16x16x32_bf16 (bf16 inputs, fp32 accum).
// Double-buffered LDS + register-staged b128 copies: one barrier per K-step,
// next slab's global loads overlap the current slab's WMMAs.
// ---------------------------------------------------------------------------

typedef __bf16 bf16;
typedef __attribute__((ext_vector_type(16))) bf16  v16bf;
typedef __attribute__((ext_vector_type(8)))  bf16  v8bfv;
typedef __attribute__((ext_vector_type(8)))  float v8f;

#define WMMA_BF16(a, b, c) \
  __builtin_amdgcn_wmma_f32_16x16x32_bf16(false, (a), false, (b), (short)0, (c), false, false)

// ---------------------------------------------------------------- conversion
__global__ void ugca_cvt_bf16(const float* __restrict__ s, bf16* __restrict__ d, int n) {
  int i = blockIdx.x * blockDim.x + threadIdx.x;
  if (i < n) d[i] = (bf16)s[i];
}

// ---------------------------------------------------------------- tiled GEMM
// C[M,N] = A[M,K] @ W[K,N] + bias, epi: 0 = bias->f32, 1 = bias->bf16,
// 2 = bias+SiLU->bf16.  Block = 256 threads = 8 waves (2x4); block tile
// 32x128, wave tile 16x32, K-step 64 -> 4 wmma per wave per step with both
// A fragments reused across two column subtiles.  Double-buffered LDS.
__global__ __launch_bounds__(256)
void ugca_gemm(const bf16* __restrict__ A, const bf16* __restrict__ W,
               const float* __restrict__ bias, float* __restrict__ outF,
               bf16* __restrict__ outB, int M, int N, int K, int epi) {
  __shared__ alignas(16) bf16 sA[2][32 * 64];    // sA[m][k]
  __shared__ alignas(16) bf16 sB[2][128 * 68];   // sB[n][k], padded stride 68

  const int t    = threadIdx.x;
  const int w    = t >> 5;
  const int lane = t & 31;
  const int wr   = w & 1;        // wave row (2)
  const int wc   = w >> 1;       // wave col (4), 32 columns each
  const int mbase = blockIdx.y * 32;
  const int nbase = blockIdx.x * 128;
  const int half  = lane >> 4;
  const int lm    = lane & 15;

  v8f acc0, acc1;
#pragma unroll
  for (int i = 0; i < 8; ++i) { acc0[i] = 0.f; acc1[i] = 0.f; }

  const int arow  = wr * 16 + lm;
  const int bcol0 = wc * 32 + lm;        // first 16-col subtile
  const int bcol1 = bcol0 + 16;          // second 16-col subtile
  const int am    = t >> 3;              // A staging: row per thread
  const int ak8   = (t & 7) * 8;         // A staging: 8-elem chunk

  int kkv[4], n8v[4];
#pragma unroll
  for (int v = 0; v < 4; ++v) {
    int vid = v * 256 + t;               // 0..1023
    kkv[v] = vid >> 4;                   // 0..63
    n8v[v] = (vid & 15) * 8;             // 0..120
  }

  // --- prologue: stage K-slab 0 into registers (clause of b128 loads) ---
  v8bfv aReg = *(const v8bfv*)&A[(size_t)(mbase + am) * K + ak8];
  v8bfv wReg[4];
#pragma unroll
  for (int v = 0; v < 4; ++v)
    wReg[v] = *(const v8bfv*)&W[(size_t)kkv[v] * N + nbase + n8v[v]];

  int cur = 0;
  for (int kb = 0; kb < K; kb += 64) {
    // --- commit staged registers to LDS buffer `cur` ---
    bf16* sAc = sA[cur];
    bf16* sBc = sB[cur];
    *(v8bfv*)&sAc[am * 64 + ak8] = aReg;
#pragma unroll
    for (int v = 0; v < 4; ++v)
#pragma unroll
      for (int i = 0; i < 8; ++i) sBc[(n8v[v] + i) * 68 + kkv[v]] = wReg[v][i];
    __syncthreads();

    // --- issue next slab's global loads; they overlap the WMMAs below ---
    if (kb + 64 < K) {
      aReg = *(const v8bfv*)&A[(size_t)(mbase + am) * K + kb + 64 + ak8];
#pragma unroll
      for (int v = 0; v < 4; ++v)
        wReg[v] = *(const v8bfv*)&W[(size_t)(kb + 64 + kkv[v]) * N + nbase + n8v[v]];
      if (kb + 128 < K)  // two steps ahead -> global_prefetch_b8
        __builtin_prefetch(&W[(size_t)(kb + 128 + kkv[0]) * N + nbase + n8v[0]], 0, 1);
    }

    // A fragments: lane row arow, k = j + 16*half (+32)
    v16bf a0, a1, b00, b01, b10, b11;
#pragma unroll
    for (int j = 0; j < 8; ++j) {
      a0[j]     = sAc[arow * 64 + 16 * half + j];
      a0[j + 8] = sAc[arow * 64 + 8 + 16 * half + j];
      a1[j]     = sAc[arow * 64 + 32 + 16 * half + j];
      a1[j + 8] = sAc[arow * 64 + 40 + 16 * half + j];
    }
    // B fragments: lane col bcol, k = j + 16*half (+32); contiguous in sB
#pragma unroll
    for (int j = 0; j < 16; ++j) {
      b00[j] = sBc[bcol0 * 68 + 16 * half + j];
      b01[j] = sBc[bcol0 * 68 + 32 + 16 * half + j];
      b10[j] = sBc[bcol1 * 68 + 16 * half + j];
      b11[j] = sBc[bcol1 * 68 + 32 + 16 * half + j];
    }
    acc0 = WMMA_BF16(a0, b00, acc0);
    acc1 = WMMA_BF16(a0, b10, acc1);   // reuses a0 from registers
    acc0 = WMMA_BF16(a1, b01, acc0);
    acc1 = WMMA_BF16(a1, b11, acc1);   // reuses a1 from registers
    cur ^= 1;                          // next store goes to the other buffer
  }

  // epilogue: C layout lane->col lm, VGPR v -> row v + 8*half
  const int col0 = nbase + wc * 32 + lm;
  const int col1 = col0 + 16;
  const float bv0 = bias[col0];
  const float bv1 = bias[col1];
#pragma unroll
  for (int v = 0; v < 8; ++v) {
    int row = mbase + wr * 16 + v + 8 * half;
    float y0 = acc0[v] + bv0;
    float y1 = acc1[v] + bv1;
    if (epi == 2) {
      y0 = y0 / (1.f + __expf(-y0));   // SiLU
      y1 = y1 / (1.f + __expf(-y1));
    }
    size_t off0 = (size_t)row * N + col0;
    size_t off1 = (size_t)row * N + col1;
    if (epi == 0) { outF[off0] = y0; outF[off1] = y1; }
    else          { outB[off0] = (bf16)y0; outB[off1] = (bf16)y1; }
  }
}

// --------------------------------------------------------- flash attention
// Block = 128 threads = 4 waves, all sharing one (batch, head); each wave
// owns a 16-row M tile.  K/V 32-key tiles are double-buffered in LDS with
// register-staged b128 copies: one barrier per step; next tile's global
// loads overlap this tile's 8 WMMAs + softmax.  S^T = K @ Q^T via WMMA so
// softmaxed P lands exactly in the A-fragment layout for O += P @ V.
__global__ __launch_bounds__(128)
void ugca_attn(const bf16* __restrict__ Q, const bf16* __restrict__ Kmat,
               const bf16* __restrict__ Vmat, const float* __restrict__ logb,
               bf16* __restrict__ Out, int Tq, int Tk) {
  __shared__ alignas(16) bf16 sK[2][32 * 72];   // [n][k], padded stride
  __shared__ alignas(16) bf16 sV[2][32 * 72];   // [n][dh], padded stride

  const int t    = threadIdx.x;
  const int wave = t >> 5;
  const int lane = t & 31;
  const int mtile = blockIdx.x * 4 + wave;
  const int h  = blockIdx.y;
  const int bz = blockIdx.z;
  const int half = lane >> 4;
  const int lm   = lane & 15;

  // Q^T as B operand: lane -> column m = lm ; element j -> k = (j&7) + 8*half + 16*(j>>3)
  const bf16* qrow = Q + (size_t)(bz * Tq + mtile * 16 + lm) * 512 + h * 64;
  v16bf bq0, bq1;
#pragma unroll
  for (int j = 0; j < 16; ++j) {
    bq0[j] = qrow[(j & 7) + 8 * half + 16 * (j >> 3)];
    bq1[j] = qrow[32 + (j & 7) + 8 * half + 16 * (j >> 3)];
  }

  // staging geometry: thread owns 2 8-elem chunks per tile
  int nst[2], c8st[2];
#pragma unroll
  for (int v = 0; v < 2; ++v) {
    int vid = t * 2 + v;               // 0..255
    nst[v]  = vid >> 3;                // 0..31
    c8st[v] = (vid & 7) * 8;           // 0..56
  }

  // --- prologue: stage tile nb=0 into registers ---
  v8bfv kReg[2], vReg[2];
#pragma unroll
  for (int v = 0; v < 2; ++v) {
    size_t g = (size_t)(bz * Tk + nst[v]) * 512 + h * 64 + c8st[v];
    kReg[v] = *(const v8bfv*)&Kmat[g];
    vReg[v] = *(const v8bfv*)&Vmat[g];
  }

  float m_i = -1e30f, l_i = 0.f;
  v8f o0, o1, o2, o3;
#pragma unroll
  for (int v = 0; v < 8; ++v) { o0[v] = 0.f; o1[v] = 0.f; o2[v] = 0.f; o3[v] = 0.f; }

  int cur = 0;
  for (int nb = 0; nb < Tk; nb += 32) {
    // --- commit staged registers to LDS buffer `cur` ---
    bf16* sKc = sK[cur];
    bf16* sVc = sV[cur];
#pragma unroll
    for (int v = 0; v < 2; ++v) {
      *(v8bfv*)&sKc[nst[v] * 72 + c8st[v]] = kReg[v];
      *(v8bfv*)&sVc[nst[v] * 72 + c8st[v]] = vReg[v];
    }
    __syncthreads();

    // --- issue next tile's global loads; overlap the compute below ---
    if (nb + 32 < Tk) {
#pragma unroll
      for (int v = 0; v < 2; ++v) {
        size_t g = (size_t)(bz * Tk + nb + 32 + nst[v]) * 512 + h * 64 + c8st[v];
        kReg[v] = *(const v8bfv*)&Kmat[g];
        vReg[v] = *(const v8bfv*)&Vmat[g];
      }
    }

    // K rows as A operand from LDS (vectorized ds loads)
    const bf16* kr0 = &sKc[lm * 72];
    const bf16* kr1 = &sKc[(16 + lm) * 72];
    v16bf ak0a, ak0b, ak1a, ak1b;
#pragma unroll
    for (int j = 0; j < 8; ++j) {
      int k0 = 8 * half + j;
      ak0a[j] = kr0[k0];      ak0a[j + 8] = kr0[16 + k0];
      ak0b[j] = kr0[32 + k0]; ak0b[j + 8] = kr0[48 + k0];
      ak1a[j] = kr1[k0];      ak1a[j + 8] = kr1[16 + k0];
      ak1b[j] = kr1[32 + k0]; ak1b[j + 8] = kr1[48 + k0];
    }
    v8f st0, st1;
#pragma unroll
    for (int v = 0; v < 8; ++v) { st0[v] = 0.f; st1[v] = 0.f; }
    st0 = WMMA_BF16(ak0a, bq0, st0);
    st0 = WMMA_BF16(ak0b, bq1, st0);
    st1 = WMMA_BF16(ak1a, bq0, st1);
    st1 = WMMA_BF16(ak1b, bq1, st1);

    // scale + per-key log bias (C row = key v + 8*half)
    float p0[8], p1[8], lmax = -1e30f;
#pragma unroll
    for (int v = 0; v < 8; ++v) {
      float s0 = st0[v] * 0.125f + logb[bz * Tk + nb + v + 8 * half];
      float s1 = st1[v] * 0.125f + logb[bz * Tk + nb + 16 + v + 8 * half];
      p0[v] = s0; p1[v] = s1;
      lmax = fmaxf(lmax, fmaxf(s0, s1));
    }
    lmax = fmaxf(lmax, __shfl_xor(lmax, 16, 32));   // combine key halves
    float newm  = fmaxf(m_i, lmax);
    float alpha = __expf(m_i - newm);
    float lsum  = 0.f;
#pragma unroll
    for (int v = 0; v < 8; ++v) {
      p0[v] = __expf(p0[v] - newm);
      p1[v] = __expf(p1[v] - newm);
      lsum += p0[v] + p1[v];
    }
    lsum += __shfl_xor(lsum, 16, 32);
    l_i = l_i * alpha + lsum;
    m_i = newm;

    // rescale O: row m of O lives at C row v+8*half; stats for m at lane m
#pragma unroll
    for (int v = 0; v < 8; ++v) {
      float al = __shfl(alpha, v + 8 * half, 32);
      o0[v] *= al; o1[v] *= al; o2[v] *= al; o3[v] *= al;
    }

    // P as A operand: element j maps straight to our own p0/p1 values.
    v16bf ap;
#pragma unroll
    for (int j = 0; j < 8; ++j) { ap[j] = (bf16)p0[j]; ap[j + 8] = (bf16)p1[j]; }

    // V as B operand from LDS: lane -> dh col lm + 16*t ; k = (j&7) + 8*half + 16*(j>>3)
    v16bf bv;
#pragma unroll
    for (int j = 0; j < 16; ++j) bv[j] = sVc[((j & 7) + 8 * half + 16 * (j >> 3)) * 72 + lm];
    o0 = WMMA_BF16(ap, bv, o0);
#pragma unroll
    for (int j = 0; j < 16; ++j) bv[j] = sVc[((j & 7) + 8 * half + 16 * (j >> 3)) * 72 + 16 + lm];
    o1 = WMMA_BF16(ap, bv, o1);
#pragma unroll
    for (int j = 0; j < 16; ++j) bv[j] = sVc[((j & 7) + 8 * half + 16 * (j >> 3)) * 72 + 32 + lm];
    o2 = WMMA_BF16(ap, bv, o2);
#pragma unroll
    for (int j = 0; j < 16; ++j) bv[j] = sVc[((j & 7) + 8 * half + 16 * (j >> 3)) * 72 + 48 + lm];
    o3 = WMMA_BF16(ap, bv, o3);
    cur ^= 1;                          // next store goes to the other buffer
  }

  bf16* orow = Out + (size_t)(bz * Tq + mtile * 16) * 512 + h * 64;
#pragma unroll
  for (int v = 0; v < 8; ++v) {
    float linv = 1.f / __shfl(l_i, v + 8 * half, 32);
    int rr = v + 8 * half;
    orow[(size_t)rr * 512 + lm]      = (bf16)(o0[v] * linv);
    orow[(size_t)rr * 512 + 16 + lm] = (bf16)(o1[v] * linv);
    orow[(size_t)rr * 512 + 32 + lm] = (bf16)(o2[v] * linv);
    orow[(size_t)rr * 512 + 48 + lm] = (bf16)(o3[v] * linv);
  }
}

// ------------------------------------------------------------- layer norm
// One wave per 512-wide row; optional fp32 residual; fp32 or bf16 out.
__global__ __launch_bounds__(256)
void ugca_ln(const float* __restrict__ x, const float* __restrict__ resid,
             const float* __restrict__ g, const float* __restrict__ b,
             float* __restrict__ outF, bf16* __restrict__ outB, int rows) {
  int wave = threadIdx.x >> 5, lane = threadIdx.x & 31;
  int row = blockIdx.x * 8 + wave;
  if (row >= rows) return;
  const float* xr = x + (size_t)row * 512;
  const float* rr = resid ? resid + (size_t)row * 512 : nullptr;
  float vals[16], s = 0.f, s2 = 0.f;
#pragma unroll
  for (int i = 0; i < 16; ++i) {
    float v = xr[lane + 32 * i];
    if (rr) v += rr[lane + 32 * i];
    vals[i] = v; s += v; s2 += v * v;
  }
#pragma unroll
  for (int o = 16; o > 0; o >>= 1) { s += __shfl_xor(s, o, 32); s2 += __shfl_xor(s2, o, 32); }
  float mean = s * (1.f / 512.f);
  float var  = s2 * (1.f / 512.f) - mean * mean;
  float inv  = rsqrtf(var + 1e-5f);
#pragma unroll
  for (int i = 0; i < 16; ++i) {
    int c = lane + 32 * i;
    float y = (vals[i] - mean) * inv * g[c] + b[c];
    if (outF) outF[(size_t)row * 512 + c] = y;
    if (outB) outB[(size_t)row * 512 + c] = (bf16)y;
  }
}

// -------------------------------------------------------- evidential tail
__device__ __forceinline__ float ugca_softplus(float x) {
  return (x > 20.f) ? x : log1pf(__expf(x));
}

__global__ __launch_bounds__(256)
void ugca_evi_tail(const bf16* __restrict__ Hh, const float* __restrict__ w2,
                   const float* __restrict__ b2, const int* __restrict__ kptr,
                   float* __restrict__ gout, float* __restrict__ logb, int T) {
  int wave = threadIdx.x >> 5, lane = threadIdx.x & 31;
  int tok = blockIdx.x * 8 + wave;
  if (tok >= T) return;
  const bf16* hr = Hh + (size_t)tok * 512;
  float a0 = 0.f, a1 = 0.f, a2 = 0.f, a3 = 0.f;
  for (int j = lane; j < 512; j += 32) {
    float hv = (float)hr[j];
    a0 += hv * w2[j * 4 + 0];
    a1 += hv * w2[j * 4 + 1];
    a2 += hv * w2[j * 4 + 2];
    a3 += hv * w2[j * 4 + 3];
  }
#pragma unroll
  for (int o = 16; o > 0; o >>= 1) {
    a0 += __shfl_xor(a0, o, 32); a1 += __shfl_xor(a1, o, 32);
    a2 += __shfl_xor(a2, o, 32); a3 += __shfl_xor(a3, o, 32);
  }
  if (lane == 0) {
    float nu    = ugca_softplus(a1 + b2[1]) + 1e-6f;
    float alpha = ugca_softplus(a2 + b2[2]) + 1.f;
    float beta  = ugca_softplus(a3 + b2[3]) + 1e-6f;
    float sigma2 = beta / (nu * (alpha - 1.f));
    float kk = (float)kptr[0];
    float g = __expf(-kk * sigma2);
    g = fminf(fmaxf(g, 0.05f), 1.f);
    gout[tok] = g;
    logb[tok] = logf(g + 1e-12f);
  }
}

// ---------------------------------------------------------------------------
// Host orchestration
// ---------------------------------------------------------------------------
extern "C" void kernel_launch(void* const* d_in, const int* in_sizes, int n_in,
                              void* d_out, int out_size, void* d_ws, size_t ws_size,
                              hipStream_t stream) {
  (void)in_sizes; (void)n_in; (void)out_size; (void)ws_size;

  // ---- inputs (setup_inputs order) ----
  const float* Hd = (const float*)d_in[0];
  const float* Hp = (const float*)d_in[1];
  const int*   kp = (const int*)d_in[2];
  const float* evid_w1 = (const float*)d_in[3];  const float* evid_b1 = (const float*)d_in[4];
  const float* evid_w2 = (const float*)d_in[5];  const float* evid_b2 = (const float*)d_in[6];
  const float* evip_w1 = (const float*)d_in[7];  const float* evip_b1 = (const float*)d_in[8];
  const float* evip_w2 = (const float*)d_in[9];  const float* evip_b2 = (const float*)d_in[10];
  const float* p2d_wq = (const float*)d_in[11]; const float* p2d_wk = (const float*)d_in[12];
  const float* p2d_wv = (const float*)d_in[13]; const float* p2d_wo = (const float*)d_in[14];
  const float* p2d_bq = (const float*)d_in[15]; const float* p2d_bk = (const float*)d_in[16];
  const float* p2d_bv = (const float*)d_in[17]; const float* p2d_bo = (const float*)d_in[18];
  const float* p2d_lng = (const float*)d_in[19]; const float* p2d_lnb = (const float*)d_in[20];
  const float* d2p_wq = (const float*)d_in[21]; const float* d2p_wk = (const float*)d_in[22];
  const float* d2p_wv = (const float*)d_in[23]; const float* d2p_wo = (const float*)d_in[24];
  const float* d2p_bq = (const float*)d_in[25]; const float* d2p_bk = (const float*)d_in[26];
  const float* d2p_bv = (const float*)d_in[27]; const float* d2p_bo = (const float*)d_in[28];
  const float* d2p_lng = (const float*)d_in[29]; const float* d2p_lnb = (const float*)d_in[30];
  const float* ffnd_w1 = (const float*)d_in[31]; const float* ffnd_b1 = (const float*)d_in[32];
  const float* ffnd_w2 = (const float*)d_in[33]; const float* ffnd_b2 = (const float*)d_in[34];
  const float* ffnd_lng = (const float*)d_in[35]; const float* ffnd_lnb = (const float*)d_in[36];
  const float* ffnp_w1 = (const float*)d_in[37]; const float* ffnp_b1 = (const float*)d_in[38];
  const float* ffnp_w2 = (const float*)d_in[39]; const float* ffnp_b2 = (const float*)d_in[40];
  const float* ffnp_lng = (const float*)d_in[41]; const float* ffnp_lnb = (const float*)d_in[42];

  // ---- outputs: Hd_out, Hp_out, gd, gp concatenated ----
  float* out   = (float*)d_out;
  float* HdOut = out;                       // 16*256*512  = 2,097,152
  float* HpOut = out + 2097152;             // 16*1024*512 = 8,388,608
  float* gdOut = out + 10485760;            // 4096
  float* gpOut = out + 10489856;            // 16384

  // ---- workspace layout (bytes) ----
  char* ws = (char*)d_ws;
  constexpr size_t MB = 1048576;
  bf16*  Wb    = (bf16*)ws;                          // bf16 weight region
  bf16*  HdB   = (bf16*)(ws + 13631488);             // 4 MB
  bf16*  HpB   = (bf16*)(ws + 17825792);             // 16 MB
  float* logbD = (float*)(ws + 34603008);            // 16 KB
  float* logbP = (float*)(ws + 34619392);            // 64 KB
  char*  S     = ws + 34684928;                      // reusable scratch

  // weight element offsets inside Wb
  constexpr size_t O_EW1D = 0,        O_EW1P = 262144;
  constexpr size_t O_PWQ = 524288,  O_PWK = 786432,  O_PWV = 1048576, O_PWO = 1310720;
  constexpr size_t O_DWQ = 1572864, O_DWK = 1835008, O_DWV = 2097152, O_DWO = 2359296;
  constexpr size_t O_FD1 = 2621440, O_FD2 = 3670016, O_FP1 = 4718592, O_FP2 = 5767168;

  auto cvt = [&](const float* src, bf16* dst, int n) {
    ugca_cvt_bf16<<<dim3((n + 255) / 256), dim3(256), 0, stream>>>(src, dst, n);
  };
  auto gemm = [&](const bf16* A, const bf16* W, const float* bias, float* oF, bf16* oB,
                  int M, int N, int K, int epi) {
    ugca_gemm<<<dim3(N / 128, M / 32), dim3(256), 0, stream>>>(A, W, bias, oF, oB, M, N, K, epi);
  };
  auto lnorm = [&](const float* x, const float* resid, const float* g, const float* b,
                   float* oF, bf16* oB, int rows) {
    ugca_ln<<<dim3(rows / 8), dim3(256), 0, stream>>>(x, resid, g, b, oF, oB, rows);
  };

  // ---- 0) convert activations + weights to bf16 ----
  cvt(Hd, HdB, 2097152);
  cvt(Hp, HpB, 8388608);
  cvt(evid_w1, Wb + O_EW1D, 262144);  cvt(evip_w1, Wb + O_EW1P, 262144);
  cvt(p2d_wq, Wb + O_PWQ, 262144);    cvt(p2d_wk, Wb + O_PWK, 262144);
  cvt(p2d_wv, Wb + O_PWV, 262144);    cvt(p2d_wo, Wb + O_PWO, 262144);
  cvt(d2p_wq, Wb + O_DWQ, 262144);    cvt(d2p_wk, Wb + O_DWK, 262144);
  cvt(d2p_wv, Wb + O_DWV, 262144);    cvt(d2p_wo, Wb + O_DWO, 262144);
  cvt(ffnd_w1, Wb + O_FD1, 1048576);  cvt(ffnd_w2, Wb + O_FD2, 1048576);
  cvt(ffnp_w1, Wb + O_FP1, 1048576);  cvt(ffnp_w2, Wb + O_FP2, 1048576);

  // ---- 1) evidential gates ----
  {
    bf16* hEd = (bf16*)S;              // 4 MB
    bf16* hEp = (bf16*)(S + 4 * MB);   // 16 MB
    gemm(HdB, Wb + O_EW1D, evid_b1, nullptr, hEd, 4096, 512, 512, 2);   // SiLU
    gemm(HpB, Wb + O_EW1P, evip_b1, nullptr, hEp, 16384, 512, 512, 2);
    ugca_evi_tail<<<dim3(512),  dim3(256), 0, stream>>>(hEd, evid_w2, evid_b2, kp, gdOut, logbD, 4096);
    ugca_evi_tail<<<dim3(2048), dim3(256), 0, stream>>>(hEp, evip_w2, evip_b2, kp, gpOut, logbP, 16384);
  }

  bf16* Hp2 = (bf16*)(S + 48 * MB);    // persistent until FFN-p
  bf16* Hd2 = (bf16*)(S + 64 * MB);    // persistent until FFN-d

  // ---- 2) p2d cross attention: Q=Hp, KV=Hd, bias=log(gd) ----
  {
    bf16* qb  = (bf16*)S;              // 16 MB
    bf16* kb  = (bf16*)(S + 16 * MB);  // 4 MB
    bf16* vb  = (bf16*)(S + 20 * MB);  // 4 MB
    bf16* ctx = (bf16*)(S + 24 * MB);  // 16 MB
    float* pj = (float*)(S + 40 * MB); // 8 MB (chunked proj)
    gemm(HpB, Wb + O_PWQ, p2d_bq, nullptr, qb, 16384, 512, 512, 1);
    gemm(HdB, Wb + O_PWK, p2d_bk, nullptr, kb, 4096, 512, 512, 1);
    gemm(HdB, Wb + O_PWV, p2d_bv, nullptr, vb, 4096, 512, 512, 1);
    ugca_attn<<<dim3(16, 8, 16), dim3(128), 0, stream>>>(qb, kb, vb, logbD, ctx, 1024, 256);
    for (int c = 0; c < 4; ++c) {
      size_t r0 = (size_t)c * 4096;
      gemm(ctx + r0 * 512, Wb + O_PWO, p2d_bo, pj, nullptr, 4096, 512, 512, 0);
      lnorm(pj, Hp + r0 * 512, p2d_lng, p2d_lnb, nullptr, Hp2 + r0 * 512, 4096);
    }
  }

  // ---- 3) d2p cross attention: Q=Hd, KV=Hp, bias=log(gp) ----
  {
    bf16* qb  = (bf16*)S;              // 4 MB
    bf16* kb  = (bf16*)(S + 4 * MB);   // 16 MB
    bf16* vb  = (bf16*)(S + 20 * MB);  // 16 MB
    bf16* ctx = (bf16*)(S + 36 * MB);  // 4 MB
    float* pj = (float*)(S + 40 * MB); // 8 MB
    gemm(HdB, Wb + O_DWQ, d2p_bq, nullptr, qb, 4096, 512, 512, 1);
    gemm(HpB, Wb + O_DWK, d2p_bk, nullptr, kb, 16384, 512, 512, 1);
    gemm(HpB, Wb + O_DWV, d2p_bv, nullptr, vb, 16384, 512, 512, 1);
    ugca_attn<<<dim3(4, 8, 16), dim3(128), 0, stream>>>(qb, kb, vb, logbP, ctx, 256, 1024);
    gemm(ctx, Wb + O_DWO, d2p_bo, pj, nullptr, 4096, 512, 512, 0);
    lnorm(pj, Hd, d2p_lng, d2p_lnb, nullptr, Hd2, 4096);
  }

  // ---- 4) FFNs ----
  {
    bf16*  hf = (bf16*)S;              // 16 MB
    float* yb = (float*)(S + 16 * MB); // 8 MB
    // FFN d
    gemm(Hd2, Wb + O_FD1, ffnd_b1, nullptr, hf, 4096, 2048, 512, 2);
    gemm(hf, Wb + O_FD2, ffnd_b2, yb, nullptr, 4096, 512, 2048, 0);
    lnorm(yb, nullptr, ffnd_lng, ffnd_lnb, HdOut, nullptr, 4096);
    // FFN p (chunked over 4x4096 tokens to bound scratch)
    for (int c = 0; c < 4; ++c) {
      size_t r0 = (size_t)c * 4096;
      gemm(Hp2 + r0 * 512, Wb + O_FP1, ffnp_b1, nullptr, hf, 4096, 2048, 512, 2);
      gemm(hf, Wb + O_FP2, ffnp_b2, yb, nullptr, 4096, 512, 2048, 0);
      lnorm(yb, nullptr, ffnp_lng, ffnp_lnb, HpOut + r0 * 512, nullptr, 4096);
    }
  }
}
